// _MockGPT2Attention_71519795413041
// MI455X (gfx1250) — matrix operand order repaired and checked by reading, not compiled, not run on hardware
//
#include <hip/hip_runtime.h>

// ---------------------------------------------------------------------------
// MockGPT2 attention for MI455X (gfx1250, wave32, WMMA bf16)
// B=4, S=2048, D=2048, H=16, HD=128
// Round 4: keep the async-to-LDS double-buffered WMMA pipeline (verified in
// disasm); fix the weight-transpose kernel to an LDS-tiled transpose so both
// global read and write are coalesced (removes ~2 GB of HBM read
// amplification), and vectorize the activation convert to b128 stores.
// ---------------------------------------------------------------------------

#define Bc   4
#define Sc   2048
#define Dc   2048
#define Hc   16
#define HDc  128
#define Mc   (Bc * Sc)                // 8192 rows
#define NX   ((size_t)Bc * Sc * Dc)   // 16,777,216 elements
#define NW   ((size_t)Dc * Dc)        // 4,194,304 elements

typedef __attribute__((ext_vector_type(16))) __bf16 v16bf;
typedef __attribute__((ext_vector_type(8)))  float  v8f;
typedef __attribute__((ext_vector_type(4)))  int    v4i;

#if __has_builtin(__builtin_amdgcn_global_load_async_to_lds_b128)
#define HAVE_ASYNC_LDS 1
#else
#define HAVE_ASYNC_LDS 0
#endif

union Frag {
  v16bf v;
  uint4 q[2];
};
static_assert(sizeof(Frag) == 32, "frag size");

__device__ __forceinline__ v8f vzero() {
  v8f z;
#pragma unroll
  for (int i = 0; i < 8; ++i) z[i] = 0.0f;
  return z;
}

__device__ __forceinline__ v8f wmma_bf16(const Frag& a, const Frag& b, v8f c) {
  // D = A(16x32 bf16) * B(32x16 bf16) + C(16x16 f32)
  return __builtin_amdgcn_wmma_f32_16x16x32_bf16(
      /*neg_a=*/false, a.v, /*neg_b=*/false, b.v,
      /*c_mod=*/(short)0, c, /*reuse_a=*/false, /*reuse_b=*/false);
}

// 16-byte global -> LDS copy. Async (ASYNCcnt) when the toolchain exposes the
// gfx1250 builtin, else classic VGPR round-trip.
__device__ __forceinline__ void lds_copy16(__bf16* lds, const __bf16* g) {
#if HAVE_ASYNC_LDS
  __builtin_amdgcn_global_load_async_to_lds_b128(
      (__attribute__((address_space(1))) v4i*)(g),
      (__attribute__((address_space(3))) v4i*)(lds),
      /*offset=*/0, /*cpol=*/0);
#else
  *(uint4*)lds = *(const uint4*)g;
#endif
}

template <int N>
__device__ __forceinline__ void wait_async() {
#if HAVE_ASYNC_LDS
# if __has_builtin(__builtin_amdgcn_s_wait_asynccnt)
  __builtin_amdgcn_s_wait_asynccnt(N);
# else
  asm volatile("s_wait_asynccnt %0" :: "n"(N) : "memory");
# endif
#endif
}

// ---------------------------------------------------------------------------
// f32 -> bf16 elementwise convert (8 elems / thread: 2x b128 read, 1x b128
// write of 8 packed bf16)
// ---------------------------------------------------------------------------
__global__ __launch_bounds__(256) void k_cvt(const float* __restrict__ s,
                                             __bf16* __restrict__ d) {
  size_t i = ((size_t)blockIdx.x * 256 + threadIdx.x) * 8;
  float4 a = *(const float4*)&s[i];
  float4 b = *(const float4*)&s[i + 4];
  __bf16 o[8];
  o[0] = (__bf16)a.x; o[1] = (__bf16)a.y; o[2] = (__bf16)a.z; o[3] = (__bf16)a.w;
  o[4] = (__bf16)b.x; o[5] = (__bf16)b.y; o[6] = (__bf16)b.z; o[7] = (__bf16)b.w;
  *(uint4*)&d[i] = *(const uint4*)&o[0];
}

// ---------------------------------------------------------------------------
// f32 W[K=2048][N=2048] -> bf16 Wt[N][K], LDS-tiled transpose.
// 32x32 f32 tile per block (256 threads = 32x8), padded LDS row (33) to avoid
// bank conflicts. Both global read and global write are coalesced.
// ---------------------------------------------------------------------------
__global__ __launch_bounds__(256) void k_cvt_t(const float* __restrict__ W,
                                               __bf16* __restrict__ Wt) {
  __shared__ float tile[32][33];
  const int tx = threadIdx.x & 31;          // fast dim
  const int ty = threadIdx.x >> 5;          // 8 rows per pass
  const int k0 = blockIdx.y * 32;
  const int n0 = blockIdx.x * 32;

#pragma unroll
  for (int i = 0; i < 32; i += 8)           // read W rows (k-major), tx = n
    tile[ty + i][tx] = W[(size_t)(k0 + ty + i) * Dc + n0 + tx];
  __syncthreads();

#pragma unroll
  for (int i = 0; i < 32; i += 8)           // write Wt rows (n-major), tx = k
    Wt[(size_t)(n0 + ty + i) * Dc + k0 + tx] = (__bf16)tile[tx][ty + i];
}

// ---------------------------------------------------------------------------
// WMMA GEMM: C[8192 x 2048] = A[8192 x 2048] * B[2048 x 2048]
//   A row-major bf16, B given as Bt[N][K] bf16.
// Block tile 128x128, BK=32, 8 waves (2 M x 4 N), wave tile 64x32 (4x2 WMMA).
// Double-buffered async staging: 4 async ops/wave/iter, s_wait_asynccnt<=4.
// mode 0/1: out bf16 in [b,h,s,hd]     (Q, K)
// mode 2  : out bf16 in [b,h,hd,s]     (V transposed for attention B-frags)
// mode 3  : out f32 row-major [row][col] (final projection)
// ---------------------------------------------------------------------------
__global__ __launch_bounds__(256) void k_gemm(const __bf16* __restrict__ A,
                                              const __bf16* __restrict__ Bt,
                                              void* __restrict__ Cout,
                                              int mode) {
  __shared__ __bf16 Ash[2][128 * 32];
  __shared__ __bf16 Bsh[2][128 * 32];

  const int tid  = threadIdx.x;
  const int lane = tid & 31, wid = tid >> 5;
  const int half = lane >> 4, l16 = lane & 15;
  const int mwave = wid & 1, nwave = wid >> 1;        // 2 x 4 wave grid
  const int blockM = blockIdx.y * 128;
  const int blockN = blockIdx.x * 128;

  v8f acc[4][2];
#pragma unroll
  for (int i = 0; i < 4; ++i)
#pragma unroll
    for (int j = 0; j < 2; ++j) acc[i][j] = vzero();

  // Stage one 128x32 A tile + 128x32 B tile into buffer `buf`.
  auto stage = [&](int buf, int k0) {
#pragma unroll
    for (int i = 0; i < 2; ++i) {
      int c = tid + i * 256;            // 0..511
      int row = c >> 2;                 // 0..127
      int off = (c & 3) * 8;            // bf16 elems, 16B chunks
      lds_copy16(&Ash[buf][row * 32 + off],
                 &A[(size_t)(blockM + row) * Dc + k0 + off]);
      lds_copy16(&Bsh[buf][row * 32 + off],
                 &Bt[(size_t)(blockN + row) * Dc + k0 + off]);
    }
  };

  stage(0, 0);
  int buf = 0;
  for (int k0 = 0; k0 < Dc; k0 += 32) {
    if (k0 + 32 < Dc) {                 // prefetch next tile, then wait for cur
      stage(buf ^ 1, k0 + 32);
      wait_async<4>();
    } else {
      wait_async<0>();
    }
    __syncthreads();

    Frag af[4], bfr[2];
#pragma unroll
    for (int i = 0; i < 4; ++i) {
      const __bf16* p = &Ash[buf][(mwave * 64 + i * 16 + l16) * 32];
      af[i].q[0] = *(const uint4*)&p[half * 8];
      af[i].q[1] = *(const uint4*)&p[16 + half * 8];
    }
#pragma unroll
    for (int j = 0; j < 2; ++j) {
      const __bf16* p = &Bsh[buf][(nwave * 32 + j * 16 + l16) * 32];
      bfr[j].q[0] = *(const uint4*)&p[half * 8];
      bfr[j].q[1] = *(const uint4*)&p[16 + half * 8];
    }
#pragma unroll
    for (int i = 0; i < 4; ++i)
#pragma unroll
      for (int j = 0; j < 2; ++j)
        acc[i][j] = wmma_bf16(af[i], bfr[j], acc[i][j]);

    __syncthreads();                    // protect buf before it is re-staged
    buf ^= 1;
  }

  // Epilogue: C/D layout -> lane N = l16, VGPR r -> M = r + 8*half
#pragma unroll
  for (int i = 0; i < 4; ++i) {
#pragma unroll
    for (int j = 0; j < 2; ++j) {
#pragma unroll
      for (int r = 0; r < 8; ++r) {
        int row = blockM + mwave * 64 + i * 16 + r + half * 8;
        int col = blockN + nwave * 32 + j * 16 + l16;
        float val = acc[i][j][r];
        if (mode == 3) {
          ((float*)Cout)[(size_t)row * Dc + col] = val;
        } else {
          int b = row >> 11, s = row & (Sc - 1);
          int h = col >> 7, hd = col & (HDc - 1);
          __bf16 bv = (__bf16)val;
          if (mode == 2) {
            ((__bf16*)Cout)[((size_t)(b * Hc + h) * HDc + hd) * Sc + s] = bv;
          } else {
            ((__bf16*)Cout)[((size_t)(b * Hc + h) * Sc + s) * HDc + hd] = bv;
          }
        }
      }
    }
  }
}

// ---------------------------------------------------------------------------
// Flash attention (causal), one (b,h) per blockIdx.y, 128 q-rows per block.
// 8 waves; each wave owns 16 q rows (independent online softmax).
// KV streamed in 32-wide tiles, double-buffered with async-to-LDS copies.
// V pre-transposed [hd][s]. Output bf16 as [b, s, h*HD + hd].
// ---------------------------------------------------------------------------
__global__ __launch_bounds__(256) void k_attn(const __bf16* __restrict__ Qg,
                                              const __bf16* __restrict__ Kg,
                                              const __bf16* __restrict__ Vt,
                                              __bf16* __restrict__ Og) {
  __shared__ __bf16 Ksh[2][32 * 128];   // [kv][hd]
  __shared__ __bf16 Vsh[2][128 * 32];   // [hd][kv]
  __shared__ __bf16 Psh[8 * 16 * 32];   // per-wave P scratch [m][kv]

  const int tid  = threadIdx.x;
  const int lane = tid & 31, w = tid >> 5;
  const int half = lane >> 4, l16 = lane & 15;
  const int bh = blockIdx.y;                  // b*16 + h
  const int q0 = blockIdx.x * 128;
  const int qb = q0 + w * 16;                 // this wave's first q row
  const size_t base = (size_t)bh * Sc * HDc;
  const __bf16* Qb = Qg + base;               // [s][hd]
  const __bf16* Kb = Kg + base;               // [s][hd]
  const __bf16* Vb = Vt + base;               // [hd][s]

  // Q fragments (A layout), kept in registers for the whole KV sweep.
  Frag qf[4];
#pragma unroll
  for (int kd = 0; kd < 4; ++kd) {
    const __bf16* p = &Qb[(size_t)(qb + l16) * HDc + kd * 32];
    qf[kd].q[0] = *(const uint4*)&p[half * 8];
    qf[kd].q[1] = *(const uint4*)&p[16 + half * 8];
  }

  v8f acc[8];
  float mi[8], li[8];
#pragma unroll
  for (int r = 0; r < 8; ++r) { mi[r] = -1e30f; li[r] = 0.0f; }
#pragma unroll
  for (int ht = 0; ht < 8; ++ht) acc[ht] = vzero();

  const float scl = 0.08838834764831845f;     // 1/sqrt(128)
  const int kv_end = q0 + 128;

  // Stage one K(32x128) + V(128x32) tile into buffer `buf`.
  auto stage_kv = [&](int buf, int kv0) {
#pragma unroll
    for (int i = 0; i < 2; ++i) {
      int c = tid + i * 256;                  // 0..511
      {
        int row = c >> 4, off = (c & 15) * 8; // 16 chunks per 128-elem row
        lds_copy16(&Ksh[buf][row * 128 + off],
                   &Kb[(size_t)(kv0 + row) * HDc + off]);
      }
      {
        int row = c >> 2, off = (c & 3) * 8;  // 4 chunks per 32-elem row
        lds_copy16(&Vsh[buf][row * 32 + off],
                   &Vb[(size_t)row * Sc + kv0 + off]);
      }
    }
  };

  stage_kv(0, 0);
  int buf = 0;
  for (int kv0 = 0; kv0 < kv_end; kv0 += 32) {
    if (kv0 + 32 < kv_end) {
      stage_kv(buf ^ 1, kv0 + 32);
      wait_async<4>();
    } else {
      wait_async<0>();
    }
    __syncthreads();

    if (kv0 <= qb + 15) {                     // causal tile skip
      // ---- scores S = Q K^T : two 16x16 tiles over kv ----
      v8f s0 = vzero(), s1 = vzero();
#pragma unroll
      for (int kd = 0; kd < 4; ++kd) {
        Frag b0, b1;
        const __bf16* p0 = &Ksh[buf][l16 * 128 + kd * 32];
        b0.q[0] = *(const uint4*)&p0[half * 8];
        b0.q[1] = *(const uint4*)&p0[16 + half * 8];
        const __bf16* p1 = &Ksh[buf][(16 + l16) * 128 + kd * 32];
        b1.q[0] = *(const uint4*)&p1[half * 8];
        b1.q[1] = *(const uint4*)&p1[16 + half * 8];
        s0 = wmma_bf16(qf[kd], b0, s0);
        s1 = wmma_bf16(qf[kd], b1, s1);
      }

      // ---- scale + causal mask (only near diagonal) ----
      const bool need_mask = (kv0 + 31 > qb);
#pragma unroll
      for (int r = 0; r < 8; ++r) {
        float a0 = s0[r] * scl, a1 = s1[r] * scl;
        if (need_mask) {
          int qg = qb + r + half * 8;
          if (kv0 + l16 > qg)      a0 = -1e30f;
          if (kv0 + 16 + l16 > qg) a1 = -1e30f;
        }
        s0[r] = a0; s1[r] = a1;
      }

      // ---- online softmax (row stats live in the 16-lane group) ----
#pragma unroll
      for (int r = 0; r < 8; ++r) {
        float rowm = fmaxf(s0[r], s1[r]);
#pragma unroll
        for (int m = 8; m >= 1; m >>= 1)
          rowm = fmaxf(rowm, __shfl_xor(rowm, m, 32));
        float mnew = fmaxf(mi[r], rowm);
        float resc = __expf(mi[r] - mnew);
        float p0 = __expf(s0[r] - mnew);
        float p1 = __expf(s1[r] - mnew);
        float rs = p0 + p1;
#pragma unroll
        for (int m = 8; m >= 1; m >>= 1)
          rs += __shfl_xor(rs, m, 32);
        li[r] = li[r] * resc + rs;
        mi[r] = mnew;
        s0[r] = p0; s1[r] = p1;
#pragma unroll
        for (int ht = 0; ht < 8; ++ht) acc[ht][r] *= resc;
      }

      // ---- P: C/D layout -> A layout via per-wave LDS scratch ----
      __bf16* Pw = &Psh[w * 16 * 32];
#pragma unroll
      for (int r = 0; r < 8; ++r) {
        int m = r + half * 8;
        Pw[m * 32 + l16]      = (__bf16)s0[r];
        Pw[m * 32 + 16 + l16] = (__bf16)s1[r];
      }
      Frag pa;
      {
        const __bf16* p = &Pw[l16 * 32];
        pa.q[0] = *(const uint4*)&p[half * 8];
        pa.q[1] = *(const uint4*)&p[16 + half * 8];
      }

      // ---- O += P V : 8 hd tiles ----
#pragma unroll
      for (int ht = 0; ht < 8; ++ht) {
        Frag bv;
        const __bf16* p = &Vsh[buf][(ht * 16 + l16) * 32];
        bv.q[0] = *(const uint4*)&p[half * 8];
        bv.q[1] = *(const uint4*)&p[16 + half * 8];
        acc[ht] = wmma_bf16(pa, bv, acc[ht]);
      }
    }
    __syncthreads();
    buf ^= 1;
  }

  // ---- normalize + store bf16 into [b, s, h*HD + hd] ----
  const int b = bh >> 4, h = bh & 15;
#pragma unroll
  for (int r = 0; r < 8; ++r) {
    float inv = 1.0f / li[r];
    int s = qb + r + half * 8;
#pragma unroll
    for (int ht = 0; ht < 8; ++ht) {
      int hd = ht * 16 + l16;
      Og[((size_t)(b * Sc + s)) * Dc + h * HDc + hd] = (__bf16)(acc[ht][r] * inv);
    }
  }
}

// ---------------------------------------------------------------------------
// Host-side launcher
// ---------------------------------------------------------------------------
extern "C" void kernel_launch(void* const* d_in, const int* in_sizes, int n_in,
                              void* d_out, int out_size, void* d_ws, size_t ws_size,
                              hipStream_t stream) {
  (void)in_sizes; (void)n_in; (void)out_size; (void)ws_size;

  const float* X  = (const float*)d_in[0];
  const float* Wq = (const float*)d_in[1];
  const float* Wk = (const float*)d_in[2];
  const float* Wv = (const float*)d_in[3];
  const float* Wo = (const float*)d_in[4];

  // Workspace carve-up (bf16 elements); ~201 MB total.
  __bf16* ws  = (__bf16*)d_ws;
  __bf16* Xb  = ws;                  // [B*S, D]
  __bf16* Wqt = Xb  + NX;            // [N, K] transposed
  __bf16* Wkt = Wqt + NW;
  __bf16* Wvt = Wkt + NW;
  __bf16* Wot = Wvt + NW;
  __bf16* Qb  = Wot + NW;            // [b,h,s,hd]
  __bf16* Kb  = Qb  + NX;            // [b,h,s,hd]
  __bf16* Vb  = Kb  + NX;            // [b,h,hd,s]
  __bf16* Ab  = Vb  + NX;            // [b*s, h*hd]

  // 1) convert inputs to bf16 (weights via LDS-tiled coalesced transpose)
  dim3 tgrid(Dc / 32, Dc / 32);      // (64, 64)
  k_cvt  <<<(unsigned)(NX / 2048), 256, 0, stream>>>(X, Xb);
  k_cvt_t<<<tgrid, 256, 0, stream>>>(Wq, Wqt);
  k_cvt_t<<<tgrid, 256, 0, stream>>>(Wk, Wkt);
  k_cvt_t<<<tgrid, 256, 0, stream>>>(Wv, Wvt);
  k_cvt_t<<<tgrid, 256, 0, stream>>>(Wo, Wot);

  // 2) QKV projections (WMMA GEMMs)
  dim3 ggrid(Dc / 128, Mc / 128);    // (16, 64)
  k_gemm<<<ggrid, 256, 0, stream>>>(Xb, Wqt, (void*)Qb, 0);
  k_gemm<<<ggrid, 256, 0, stream>>>(Xb, Wkt, (void*)Kb, 1);
  k_gemm<<<ggrid, 256, 0, stream>>>(Xb, Wvt, (void*)Vb, 2);

  // 3) causal flash attention
  dim3 agrid(Sc / 128, Bc * Hc);     // (16, 64)
  k_attn<<<agrid, 256, 0, stream>>>(Qb, Kb, Vb, Ab);

  // 4) output projection -> f32
  k_gemm<<<ggrid, 256, 0, stream>>>(Ab, Wot, d_out, 3);
}
